// xLSTM_84490596647035
// MI455X (gfx1250) — compile-verified, matrix operand order checked by reading, not run on hardware
//
#include <hip/hip_runtime.h>

// xLSTM fused forward for MI455X (gfx1250, wave32, WMMA).
// B=8, T=2048, D=256, L=8, E=64. One workgroup per (b,l) head-pair.
// Per-head weights live in LDS as bf16; gate projections for 16 timesteps are
// computed with v_wmma_f32_16x16x32_bf16; the 2048-step recurrence runs in
// fp32 with C (64x64) distributed 16 floats/thread across 256 threads.

#define B_ 8
#define T_ 2048
#define D_ 256
#define L_ 8
#define E_ 64
#define NT 16            // timesteps per tile
#define NTILES (T_ / NT) // 128
#define GROW 260         // padded gate row (floats): [o(64)|v(64)|k(64)|q(64)] + pad

typedef __bf16 v16bf __attribute__((ext_vector_type(16)));
typedef __bf16 v8bf  __attribute__((ext_vector_type(8)));
typedef float  v8f   __attribute__((ext_vector_type(8)));

// ---- dynamic LDS layout (bytes) ----
// floats: gates[2][16][GROW]=8320 | iw 256 | fw 256 | bias 256 | ifg 64 | red 256 | blin 4
#define SF_GATES 0
#define SF_IW    8320
#define SF_FW    8576
#define SF_BIAS  8832
#define SF_IFG   9088
#define SF_RED   9152
#define SF_BLIN  9408
#define SF_END   9412                       // floats
#define SB_WL    (SF_END * 4)               // 37648 : bf16 W [256 cols][256 k] = 131072 B
#define SB_XL    (SB_WL + 256 * 256 * 2)    // 168720: bf16 x [2][16][256]     = 16384 B
#define SMEM_BYTES (SB_XL + 2 * NT * D_ * 2) // 185104 B

__global__ __launch_bounds__(256) void xlstm_fused_kernel(
    const float* __restrict__ x,    // (B,T,D)
    const float* __restrict__ Wml,  // (4L,D,E)
    const float* __restrict__ bml,  // (4L,E)
    const float* __restrict__ Wlin, // (2L,D)
    const float* __restrict__ blin, // (2L,)
    float* __restrict__ out)        // h (B,L,E) then C (B,L,E,E)
{
  extern __shared__ char smem[];
  float*  sf     = (float*)smem;
  float*  gates  = sf + SF_GATES;
  float*  iw     = sf + SF_IW;
  float*  fw     = sf + SF_FW;
  float*  bias   = sf + SF_BIAS;
  float*  ifg    = sf + SF_IFG;   // [buf][which][16]
  float*  red    = sf + SF_RED;   // reduction scratch
  float*  blin_s = sf + SF_BLIN;
  __bf16* Wl     = (__bf16*)(smem + SB_WL); // [ncol=g*64+e][k=0..255]
  __bf16* xl     = (__bf16*)(smem + SB_XL); // [buf][t=0..15][k=0..255]

  const int tid  = threadIdx.x;
  const int lane = tid & 31;
  const int wave = tid >> 5;
  const int half = lane >> 4;
  const int l16  = lane & 15;
  const int pair = blockIdx.x;    // b*L + l
  const int bi   = pair >> 3;
  const int li   = pair & 7;

  // ---------------- load & pack per-head weights into LDS ----------------
  {
    const int g = tid >> 6, e = tid & 63;   // tid == ncol = g*64 + e
    const float* wsrc = Wml + (size_t)(g * L_ + li) * D_ * E_ + e;
    __bf16* wdst = Wl + tid * D_;
#pragma unroll 4
    for (int d = 0; d < D_; ++d) wdst[d] = (__bf16)wsrc[(size_t)d * E_];
    bias[tid] = bml[(g * L_ + li) * E_ + e];
    iw[tid]   = Wlin[li * D_ + tid];
    fw[tid]   = Wlin[(L_ + li) * D_ + tid];
    if (tid == 0) { blin_s[0] = blin[li]; blin_s[1] = blin[L_ + li]; }
  }

  // ---------------- x tile prefetch (registers) ----------------
  const float* xb    = x + (size_t)bi * T_ * D_;
  const int    tl    = tid >> 4;          // 0..15 timestep within tile
  const int    dbase = (tid & 15) * 16;   // 16 contiguous k per thread
  float4 xr0, xr1, xr2, xr3;

  auto load_x = [&](int tile) {
    const float* src = xb + (size_t)(tile * NT + tl) * D_ + dbase;
    xr0 = *(const float4*)(src + 0);
    xr1 = *(const float4*)(src + 4);
    xr2 = *(const float4*)(src + 8);
    xr3 = *(const float4*)(src + 12);
  };
  auto store_x = [&](int buf) {
    __bf16* dst = xl + buf * (NT * D_) + tl * D_ + dbase;
    v8bf p0 = { (__bf16)xr0.x, (__bf16)xr0.y, (__bf16)xr0.z, (__bf16)xr0.w,
                (__bf16)xr1.x, (__bf16)xr1.y, (__bf16)xr1.z, (__bf16)xr1.w };
    v8bf p1 = { (__bf16)xr2.x, (__bf16)xr2.y, (__bf16)xr2.z, (__bf16)xr2.w,
                (__bf16)xr3.x, (__bf16)xr3.y, (__bf16)xr3.z, (__bf16)xr3.w };
    *(v8bf*)dst       = p0;
    *(v8bf*)(dst + 8) = p1;
  };

  // ---------------- gate production: 16x256 = x_tile(16x256) @ W(256x256) ----------------
  auto produce = [&](int buf) {
    v8f acc0 = {};
    v8f acc1 = {};
    const __bf16* xrow  = xl + buf * (NT * D_) + l16 * D_;
    const int nb0 = wave * 32, nb1 = nb0 + 16;       // two 16-wide N tiles per wave
    const __bf16* b0row = Wl + (nb0 + l16) * D_;
    const __bf16* b1row = Wl + (nb1 + l16) * D_;
#pragma unroll
    for (int kk = 0; kk < 8; ++kk) {                 // K = 256 in steps of 32
      const int k0 = kk * 32 + (half ? 8 : 0);       // per-lane K chunk base (ISA layout)
      v8bf alo = *(const v8bf*)(xrow + k0);
      v8bf ahi = *(const v8bf*)(xrow + k0 + 16);
      v16bf a = __builtin_shufflevector(alo, ahi, 0,1,2,3,4,5,6,7,8,9,10,11,12,13,14,15);
      v8bf bl0 = *(const v8bf*)(b0row + k0);
      v8bf bh0 = *(const v8bf*)(b0row + k0 + 16);
      v16bf b0 = __builtin_shufflevector(bl0, bh0, 0,1,2,3,4,5,6,7,8,9,10,11,12,13,14,15);
      acc0 = __builtin_amdgcn_wmma_f32_16x16x32_bf16(false, a, false, b0,
                                                     (short)0, acc0, false, false);
      v8bf bl1 = *(const v8bf*)(b1row + k0);
      v8bf bh1 = *(const v8bf*)(b1row + k0 + 16);
      v16bf b1 = __builtin_shufflevector(bl1, bh1, 0,1,2,3,4,5,6,7,8,9,10,11,12,13,14,15);
      acc1 = __builtin_amdgcn_wmma_f32_16x16x32_bf16(false, a, false, b1,
                                                     (short)0, acc1, false, false);
    }
    const float bv0 = bias[nb0 + l16], bv1 = bias[nb1 + l16];
    float* gbuf = gates + buf * (NT * GROW);
#pragma unroll
    for (int r = 0; r < 8; ++r) {                    // D layout: M = r + 8*half
      const int t = r + 8 * half;
      gbuf[t * GROW + nb0 + l16] = acc0[r] + bv0;
      gbuf[t * GROW + nb1 + l16] = acc1[r] + bv1;
    }
    // linear gates i,f : 32 dot(256) products, 8 partial segments each
    {
      const int dotid = tid >> 3;                    // 0..31 = (which,t)
      const int tt = dotid & 15, which = dotid >> 4;
      const int seg = (tid & 7) * 32;
      const __bf16* xs = xl + buf * (NT * D_) + tt * D_ + seg;
      const float*  w  = (which ? fw : iw) + seg;
      float s = 0.f;
#pragma unroll 8
      for (int d = 0; d < 32; ++d) s += (float)xs[d] * w[d];
      red[tid] = s;
    }
    __syncthreads();
    if (tid < 32) {
      const int tt = tid & 15, which = tid >> 4;
      float s = blin_s[which];
#pragma unroll
      for (int j = 0; j < 8; ++j) s += red[tid * 8 + j];
      ifg[buf * 32 + which * 16 + tt] = s;
    }
    __syncthreads();
  };

  // ---------------- recurrence state ----------------
  const int row  = tid >> 2;            // v-axis index i (0..63)
  const int col0 = (tid & 3) * 16;      // k-axis columns j
  float c[16];
#pragma unroll
  for (int e = 0; e < 16; ++e) c[e] = 0.f;
  float m = 0.f, n_reg = 0.f;           // m replicated in every thread; n on tid<64

  auto scan_tile = [&](int buf) {
    const float* gbase = gates + buf * (NT * GROW);
    const float* ifb   = ifg + buf * 32;
    for (int t = 0; t < NT; ++t) {      // no barriers: all shared reads are tile-static
      const float i_t = ifb[t];
      const float f_t = ifb[16 + t];
      const float m_new = fmaxf(f_t + m, i_t);
      const float i_g = __expf(i_t - m_new);
      const float f_g = __expf(f_t + m - m_new);
      m = m_new;
      const float* gr = gbase + t * GROW;
      const float iv = i_g * gr[64 + row];           // i_g * v[row]
      const float* kp = gr + 128 + col0;
      float kv[16];
      *(float4*)&kv[0]  = *(const float4*)(kp + 0);
      *(float4*)&kv[4]  = *(const float4*)(kp + 4);
      *(float4*)&kv[8]  = *(const float4*)(kp + 8);
      *(float4*)&kv[12] = *(const float4*)(kp + 12);
#pragma unroll
      for (int e = 0; e < 16; ++e) c[e] = fmaf(f_g, c[e], iv * kv[e]);
      n_reg = fmaf(f_g, n_reg, i_g * gr[128 + (tid & 63)]); // valid for tid<64
    }
  };

  // ---------------- software-pipelined main loop ----------------
  load_x(0);
  store_x(0);
  __syncthreads();
  produce(0);
  for (int j = 0; j < NTILES; ++j) {
    const int cur = j & 1, nxt = cur ^ 1;
    if (j + 1 < NTILES) load_x(j + 1);  // global loads in flight during scan
    scan_tile(cur);
    if (j + 1 < NTILES) {
      store_x(nxt);
      __syncthreads();
      produce(nxt);                     // ends with __syncthreads()
    }
  }

  // ---------------- finalize: h = sigmoid(o) * (C^T q) / max(n.q, 1) ----------------
  const int lastbuf = (NTILES - 1) & 1;
  const float* gfin = gates + lastbuf * (NT * GROW) + (NT - 1) * GROW;
  if (tid < 64) {
    red[tid]       = n_reg * gfin[192 + tid];  // n[e]*q[e]
    red[128 + tid] = 0.f;                      // h accumulator
  }
  __syncthreads();
  if (tid < 8) {
    float s = 0.f;
#pragma unroll
    for (int j = 0; j < 8; ++j) s += red[tid * 8 + j];
    red[64 + tid] = s;
  }
  {
    const float q_r = gfin[192 + row];
#pragma unroll
    for (int e = 0; e < 16; ++e) atomicAdd(&red[128 + col0 + e], c[e] * q_r); // ds_add_f32
  }
  __syncthreads();
  if (tid < 64) {
    float denom = 0.f;
#pragma unroll
    for (int j = 0; j < 8; ++j) denom += red[64 + j];
    denom = fmaxf(denom, 1.0f);
    const float ov  = gfin[tid];
    const float sig = 1.0f / (1.0f + __expf(-ov));
    out[pair * 64 + tid] = sig * red[128 + tid] / denom;
  }
  // final C (B,L,E,E)
  float* outC = out + (B_ * L_ * E_) + ((size_t)pair * 64 + row) * 64 + col0;
#pragma unroll
  for (int e4 = 0; e4 < 4; ++e4) {
    *(float4*)(outC + e4 * 4) =
        make_float4(c[e4 * 4 + 0], c[e4 * 4 + 1], c[e4 * 4 + 2], c[e4 * 4 + 3]);
  }
}

extern "C" void kernel_launch(void* const* d_in, const int* in_sizes, int n_in,
                              void* d_out, int out_size, void* d_ws, size_t ws_size,
                              hipStream_t stream) {
  (void)in_sizes; (void)n_in; (void)out_size; (void)d_ws; (void)ws_size;
  const float* x    = (const float*)d_in[0];
  const float* Wml  = (const float*)d_in[1];
  const float* bml  = (const float*)d_in[2];
  const float* Wlin = (const float*)d_in[3];
  const float* blin = (const float*)d_in[4];
  float* out = (float*)d_out;

  hipFuncSetAttribute((const void*)xlstm_fused_kernel,
                      hipFuncAttributeMaxDynamicSharedMemorySize,
                      (int)SMEM_BYTES);
  xlstm_fused_kernel<<<dim3(B_ * L_), dim3(256), SMEM_BYTES, stream>>>(
      x, Wml, bml, Wlin, blin, out);
}
